// Decoder_47571057770796
// MI455X (gfx1250) — matrix operand order repaired and checked by reading, not compiled, not run on hardware
//
#include <hip/hip_runtime.h>
#include <math.h>

// Problem constants (match reference)
#define Bb 128
#define Tt 15
#define Hh 1024
#define Vv 32000
#define H3 (3 * Hh)

// LDS tile: 64 rows x 32 bf16 K-slice, padded to 40 shorts (80B) per row so
// async 16B writes are aligned and 16-lane ds_load_b128 reads are conflict-free.
#define WT_PAD 40

typedef __bf16 bf16x16 __attribute__((ext_vector_type(16)));
typedef float  f32x8   __attribute__((ext_vector_type(8)));

union FragAB {
    bf16x16 v;
    uint4   q[2];
};

__device__ __forceinline__ unsigned short f2bf(float f) {
    unsigned u = __float_as_uint(f);
    unsigned r = (u + 0x7fffu + ((u >> 16) & 1u)) >> 16;   // round-to-nearest-even
    return (unsigned short)r;
}

// ---------------------------------------------------------------------------
// Elementwise fp32 -> bf16 cast
// ---------------------------------------------------------------------------
__global__ void cast_bf16_kernel(const float* __restrict__ src,
                                 unsigned short* __restrict__ dst, size_t n) {
    size_t i = (size_t)blockIdx.x * blockDim.x + threadIdx.x;
    if (i < n) dst[i] = f2bf(src[i]);
}

// ---------------------------------------------------------------------------
// Embedding gather + ReLU + cast, producing x_bf16 laid out [T, B, H]
// token(t,b) = (t==0) ? 0 : target[b, t-1]
// ---------------------------------------------------------------------------
__global__ void embed_relu_kernel(const int* __restrict__ tgt,
                                  const float* __restrict__ emb,
                                  unsigned short* __restrict__ xbf) {
    size_t idx = (size_t)blockIdx.x * blockDim.x + threadIdx.x;   // over T*B*H
    if (idx >= (size_t)Tt * Bb * Hh) return;
    int i = (int)(idx & (Hh - 1));
    size_t rb = idx >> 10;            // t*B + b
    int b = (int)(rb % Bb);
    int t = (int)(rb / Bb);
    int tok = (t == 0) ? 0 : tgt[b * Tt + (t - 1)];
    float v = emb[(size_t)tok * Hh + i];
    v = v > 0.f ? v : 0.f;
    xbf[idx] = f2bf(v);
}

// ---------------------------------------------------------------------------
// h0 init: fp32 copy + bf16 copy of encoder_hidden[0]
// ---------------------------------------------------------------------------
__global__ void init_h_kernel(const float* __restrict__ eh,
                              float* __restrict__ h,
                              unsigned short* __restrict__ hbf) {
    size_t i = (size_t)blockIdx.x * blockDim.x + threadIdx.x;
    if (i >= (size_t)Bb * Hh) return;
    float v = eh[i];
    h[i] = v;
    hbf[i] = f2bf(v);
}

// ---------------------------------------------------------------------------
// bf16 WMMA GEMM with async-LDS double-buffered weight staging:
//   Y[M,N] = A[M,K] * W[N,K]^T + bias[N]   (fp32 out)
//
// Block = 256 threads = 8 waves, computing a 128(M) x 64(N) macro-tile.
// Per K-step (32), the block cooperatively streams the shared 64x32 bf16 W
// tile into LDS with global_load_async_to_lds_b128 (ASYNCcnt), double
// buffered; each wave then builds B fragments from LDS (ds_load_b128) and
// A fragments from its private global rows, issuing 4 wmma per step.
// M must be a multiple of 128, N of 64, K of 32.
// ---------------------------------------------------------------------------
__global__ void __launch_bounds__(256)
gemm_bf16_wmma(const unsigned short* __restrict__ A,
               const unsigned short* __restrict__ W,
               const float* __restrict__ bias,
               float* __restrict__ Y,
               int M, int N, int K) {
    __shared__ unsigned short wtile[2][64 * WT_PAD];   // 2 x 5 KB

    const int nBlocks = N >> 6;                 // 64-wide N tiles
    const int macroM  = blockIdx.x / nBlocks;   // 128-row macro tile
    const int nBlk    = blockIdx.x % nBlocks;

    const int wave = threadIdx.x >> 5;
    const int lane = threadIdx.x & 31;
    const int half = lane >> 4;      // 0: lanes 0-15, 1: lanes 16-31
    const int l16  = lane & 15;

    const int mTile = macroM * 8 + wave;
    const int mRow  = mTile * 16 + l16;          // A row this lane streams
    const int n0    = nBlk * 64;
    const unsigned short* __restrict__ arow = A + (size_t)mRow * K;

    // Cooperative W-tile copy: thread tid covers row tid/4, 16B chunk tid%4.
    const int crow   = threadIdx.x >> 2;         // 0..63
    const int cchunk = threadIdx.x & 3;          // 0..3
    const unsigned short* __restrict__ wsrc =
        W + (size_t)(n0 + crow) * K + cchunk * 8;

    f32x8 acc[4] = {};

    // Prologue: stage K-block 0 into buffer 0.
    {
        unsigned ldsa =
            (unsigned)(unsigned long long)&wtile[0][crow * WT_PAD + cchunk * 8];
        asm volatile("global_load_async_to_lds_b128 %0, %1, off"
                     :: "v"(ldsa), "v"((unsigned long long)wsrc) : "memory");
        asm volatile("s_wait_asynccnt 0" ::: "memory");
    }
    __syncthreads();

    const int steps = K >> 5;
    for (int s = 0; s < steps; ++s) {
        const int k0  = s << 5;
        const int cur = s & 1;
        const int nxt = cur ^ 1;

        // Kick off async stage of the next K-block while we compute.
        if (s + 1 < steps) {
            unsigned ldsa = (unsigned)(unsigned long long)
                &wtile[nxt][crow * WT_PAD + cchunk * 8];
            asm volatile("global_load_async_to_lds_b128 %0, %1, off"
                         :: "v"(ldsa),
                            "v"((unsigned long long)(wsrc + k0 + 32)) : "memory");
            __builtin_prefetch(arow + k0 + 32 + half * 8, 0, 1);
        }

        // A fragment: per-wave private rows, direct from global (L2 resident).
        FragAB a;
        a.q[0] = *(const uint4*)(arow + k0 + half * 8);        // K = h*8..h*8+7
        a.q[1] = *(const uint4*)(arow + k0 + 16 + half * 8);   // K = 16+h*8..

#pragma unroll
        for (int j = 0; j < 4; ++j) {
            const unsigned short* bp =
                &wtile[cur][(j * 16 + l16) * WT_PAD + half * 8];
            FragAB bfr;
            bfr.q[0] = *(const uint4*)(bp);        // ds_load_b128
            bfr.q[1] = *(const uint4*)(bp + 16);   // ds_load_b128
            acc[j] = __builtin_amdgcn_wmma_f32_16x16x32_bf16(
                false, a.v, false, bfr.v, (short)0, acc[j], false, false);
        }

        // Own async writes done, then block-wide barrier: next buffer is
        // fully visible and everyone is done reading the current one.
        asm volatile("s_wait_asynccnt 0" ::: "memory");
        __syncthreads();
    }

    // Epilogue: bias add + fp32 store.  D VGPR r -> row half*8+r, col lane%16.
#pragma unroll
    for (int j = 0; j < 4; ++j) {
        const int n = n0 + j * 16 + l16;
        const float bv = bias ? bias[n] : 0.0f;
#pragma unroll
        for (int r = 0; r < 8; ++r) {
            const int mr = mTile * 16 + half * 8 + r;
            Y[(size_t)mr * N + n] = acc[j][r] + bv;
        }
    }
}

// ---------------------------------------------------------------------------
// GRU gate fusion for step t:
//   r = sig(gx_r+gh_r); z = sig(gx_z+gh_z); n = tanh(gx_n + r*gh_n)
//   h' = (1-z)*n + z*h
// Writes h (fp32), h_bf16 (next-step GEMM A), hs_bf16[b*T+t] (logits GEMM A).
// Biases were already folded into gx / gh by the GEMM epilogue.
// ---------------------------------------------------------------------------
__global__ void gru_gates_kernel(const float* __restrict__ gx,   // [T*B, 3H], row t*B+b
                                 const float* __restrict__ gh,   // [B, 3H]
                                 float* __restrict__ h,          // [B, H]
                                 unsigned short* __restrict__ hbf,
                                 unsigned short* __restrict__ hs, // [B*T, H]
                                 int t) {
    size_t idx = (size_t)blockIdx.x * blockDim.x + threadIdx.x;   // over B*H
    if (idx >= (size_t)Bb * Hh) return;
    const int i = (int)(idx & (Hh - 1));
    const int b = (int)(idx >> 10);

    const size_t gxb = ((size_t)t * Bb + b) * H3;
    const size_t ghb = (size_t)b * H3;

    const float r = 1.0f / (1.0f + expf(-(gx[gxb + i] + gh[ghb + i])));
    const float z = 1.0f / (1.0f + expf(-(gx[gxb + Hh + i] + gh[ghb + Hh + i])));
    const float n = tanhf(gx[gxb + 2 * Hh + i] + r * gh[ghb + 2 * Hh + i]);

    const float hn = (1.0f - z) * n + z * h[idx];
    h[idx] = hn;
    hbf[idx] = f2bf(hn);
    hs[((size_t)b * Tt + t) * Hh + i] = f2bf(hn);
}

// ---------------------------------------------------------------------------
// In-place log-softmax over rows of V=32000.  One 256-thread block per row.
// ---------------------------------------------------------------------------
__global__ void logsoftmax_kernel(float* __restrict__ logits) {
    __shared__ float red[256];
    const int row = blockIdx.x;
    float* __restrict__ p = logits + (size_t)row * Vv;

    float m = -3.402823466e38f;
    for (int i = threadIdx.x; i < Vv; i += 256) m = fmaxf(m, p[i]);
    red[threadIdx.x] = m;
    __syncthreads();
    for (int s = 128; s > 0; s >>= 1) {
        if (threadIdx.x < s)
            red[threadIdx.x] = fmaxf(red[threadIdx.x], red[threadIdx.x + s]);
        __syncthreads();
    }
    m = red[0];
    __syncthreads();

    float sum = 0.0f;
    for (int i = threadIdx.x; i < Vv; i += 256) sum += expf(p[i] - m);
    red[threadIdx.x] = sum;
    __syncthreads();
    for (int s = 128; s > 0; s >>= 1) {
        if (threadIdx.x < s) red[threadIdx.x] += red[threadIdx.x + s];
        __syncthreads();
    }
    const float lse = m + logf(red[0]);
    for (int i = threadIdx.x; i < Vv; i += 256) p[i] -= lse;
}

// ---------------------------------------------------------------------------
// Final hidden state copy into the tail of d_out
// ---------------------------------------------------------------------------
__global__ void copy_f32_kernel(const float* __restrict__ src,
                                float* __restrict__ dst, size_t n) {
    size_t i = (size_t)blockIdx.x * blockDim.x + threadIdx.x;
    if (i < n) dst[i] = src[i];
}

// ---------------------------------------------------------------------------
extern "C" void kernel_launch(void* const* d_in, const int* in_sizes, int n_in,
                              void* d_out, int out_size, void* d_ws, size_t ws_size,
                              hipStream_t stream) {
    (void)in_sizes; (void)n_in; (void)out_size; (void)ws_size;

    const float* enc_hidden = (const float*)d_in[1];   // [1,B,H]
    const int*   target     = (const int*)d_in[2];     // [B,T]
    const float* emb        = (const float*)d_in[3];   // [V,H]
    const float* w_ih       = (const float*)d_in[4];   // [3H,H]
    const float* w_hh       = (const float*)d_in[5];   // [3H,H]
    const float* b_ih       = (const float*)d_in[6];   // [3H]
    const float* b_hh       = (const float*)d_in[7];   // [3H]
    const float* out_w      = (const float*)d_in[8];   // [V,H]
    const float* out_b      = (const float*)d_in[9];   // [V]

    // ---- workspace carve-up (all 256B aligned) ----
    char* ws = (char*)d_ws;
    size_t off = 0;
    auto alloc = [&](size_t bytes) -> char* {
        char* p = ws + off;
        off += (bytes + 255) & ~(size_t)255;
        return p;
    };
    unsigned short* w_ih_bf  = (unsigned short*)alloc((size_t)H3 * Hh * 2);   // 6.3 MB
    unsigned short* w_hh_bf  = (unsigned short*)alloc((size_t)H3 * Hh * 2);   // 6.3 MB
    unsigned short* out_w_bf = (unsigned short*)alloc((size_t)Vv * Hh * 2);   // 65.5 MB
    unsigned short* x_bf     = (unsigned short*)alloc((size_t)Tt * Bb * Hh * 2);
    unsigned short* hs_bf    = (unsigned short*)alloc((size_t)Bb * Tt * Hh * 2);
    unsigned short* h_bf     = (unsigned short*)alloc((size_t)Bb * Hh * 2);
    float* gx = (float*)alloc((size_t)Tt * Bb * H3 * 4);                       // 23.6 MB
    float* gh = (float*)alloc((size_t)Bb * H3 * 4);                            // 1.6 MB
    float* h  = (float*)alloc((size_t)Bb * Hh * 4);

    float* logits = (float*)d_out;                         // [B*T, V]
    float* hT_out = (float*)d_out + (size_t)Bb * Tt * Vv;  // [B*H]

    // ---- 1. weight casts to bf16 ----
    {
        size_t n = (size_t)H3 * Hh;
        cast_bf16_kernel<<<(unsigned)((n + 255) / 256), 256, 0, stream>>>(w_ih, w_ih_bf, n);
        cast_bf16_kernel<<<(unsigned)((n + 255) / 256), 256, 0, stream>>>(w_hh, w_hh_bf, n);
        size_t no = (size_t)Vv * Hh;
        cast_bf16_kernel<<<(unsigned)((no + 255) / 256), 256, 0, stream>>>(out_w, out_w_bf, no);
    }

    // ---- 2. embed + relu -> x_bf [T,B,H]; h0 init ----
    {
        size_t n = (size_t)Tt * Bb * Hh;
        embed_relu_kernel<<<(unsigned)((n + 255) / 256), 256, 0, stream>>>(target, emb, x_bf);
        size_t nh = (size_t)Bb * Hh;
        init_h_kernel<<<(unsigned)((nh + 255) / 256), 256, 0, stream>>>(enc_hidden, h, h_bf);
    }

    // ---- 3. hoisted input GEMM: gx[T*B,3H] = x * w_ih^T + b_ih ----
    {
        int M = Tt * Bb, N = H3, K = Hh;
        int blocks = (M / 128) * (N / 64);                 // 15 * 48
        gemm_bf16_wmma<<<blocks, 256, 0, stream>>>(x_bf, w_ih_bf, b_ih, gx, M, N, K);
    }

    // ---- 4. recurrent chain: 15 x (gh GEMM + fused gates) ----
    for (int t = 0; t < Tt; ++t) {
        int M = Bb, N = H3, K = Hh;
        int blocks = (M / 128) * (N / 64);                 // 48
        gemm_bf16_wmma<<<blocks, 256, 0, stream>>>(h_bf, w_hh_bf, b_hh, gh, M, N, K);
        size_t n = (size_t)Bb * Hh;
        gru_gates_kernel<<<(unsigned)((n + 255) / 256), 256, 0, stream>>>(gx, gh, h, h_bf, hs_bf, t);
    }

    // ---- 5. output projection: logits[B*T,V] = hs * out_w^T + out_b ----
    {
        int M = Bb * Tt, N = Vv, K = Hh;
        int blocks = (M / 128) * (N / 64);                 // 15 * 500
        gemm_bf16_wmma<<<blocks, 256, 0, stream>>>(hs_bf, out_w_bf, out_b, logits, M, N, K);
    }

    // ---- 6. log-softmax in place over each of the 1920 rows ----
    logsoftmax_kernel<<<Bb * Tt, 256, 0, stream>>>(logits);

    // ---- 7. hT -> tail of d_out ----
    {
        size_t n = (size_t)Bb * Hh;
        copy_f32_kernel<<<(unsigned)((n + 255) / 256), 256, 0, stream>>>(h, hT_out, n);
    }
}
